// MomentumActorNet_18700287607234
// MI455X (gfx1250) — matrix operand order repaired and checked by reading, not compiled, not run on hardware
//
#include <hip/hip_runtime.h>
#include <hip/hip_bf16.h>

// ---------------------------------------------------------------------------
// MomentumActorNet fused inference for gfx1250 (MI455X, wave32, WMMA bf16)
// Round 3: 32 rows/block kept (2x B-fragment reuse, 4 GB L2 weight traffic),
// but each wave's 8 col-tiles split into two passes of 4 so live accumulators
// stay at 64 VGPRs -> no VGPR-MSB bank switching / mov storm in codegen.
// Second LDS activation buffer (Ho) removes the GEMM2 read/write hazard.
// ---------------------------------------------------------------------------
#define Btot    65536
#define Tlen    512
#define OBS     256
#define HID     1024
#define NEWMA   4
#define INDIM   260           // OBS + NEWMA
#define K1P     288           // INDIM padded to multiple of 32
#define XS_STRIDE 296         // K1P + 8 (16B-aligned rows, dodge bank conflicts)
#define HS      1032          // HID + 8
#define ROWS    32            // rows per block (2 x 16-row tiles)
#define LN_EPS  1e-5f

typedef __attribute__((ext_vector_type(16))) __bf16 v16bf;
typedef __attribute__((ext_vector_type(8)))  float  v8f;

union FragU { v16bf v; uint4 q[2]; };

__device__ __forceinline__ unsigned short f2bf(float f) {
    unsigned u = __float_as_uint(f);
    u += 0x7FFFu + ((u >> 16) & 1u);          // round-to-nearest-even
    return (unsigned short)(u >> 16);
}
__device__ __forceinline__ float bf2f(unsigned short h) {
    return __uint_as_float(((unsigned)h) << 16);
}

// ---------------------------------------------------------------------------
// Kernel 0: EWMA weight table w[4][512]  (fp32, tiny, runs once)
// ---------------------------------------------------------------------------
__global__ void ewma_weights_kernel(const float* __restrict__ log_hl,
                                    float* __restrict__ w) {
    int t = threadIdx.x;                       // 0..511
    if (t >= Tlen) return;
    #pragma unroll
    for (int k = 0; k < NEWMA; ++k) {
        float hl = expf(log_hl[k]);
        hl = fminf(fmaxf(hl, 1.0f), 200.0f);
        float a  = exp2f(-1.0f / hl);
        float pw = exp2f(-((float)(Tlen - 1 - t)) / hl);   // a^(T-1-t)
        float wv = (t == 0) ? pw : (1.0f - a) * pw;
        w[k * Tlen + t] = wv;
    }
}

// ---------------------------------------------------------------------------
// Kernel 1: fc1_w (1024 x 260 fp32) -> bf16 padded (1024 x 288)
// ---------------------------------------------------------------------------
__global__ void convert_w1_kernel(const float* __restrict__ w1,
                                  unsigned short* __restrict__ out) {
    int i = blockIdx.x * blockDim.x + threadIdx.x;   // 0 .. 1024*288-1
    int n = i / K1P, k = i - n * K1P;
    float v = (k < INDIM) ? w1[n * INDIM + k] : 0.0f;
    out[i] = f2bf(v);
}

// ---------------------------------------------------------------------------
// Kernel 2: fc2_w (1024 x 1024 fp32) -> bf16
// ---------------------------------------------------------------------------
__global__ void convert_w2_kernel(const float* __restrict__ w2,
                                  unsigned short* __restrict__ out) {
    int i = blockIdx.x * blockDim.x + threadIdx.x;
    out[i] = f2bf(w2[i]);
}

// ---------------------------------------------------------------------------
// Kernel 3: fully fused EWMA + pack + MLP.
// 32 rows per block, 256 threads = 8 waves. Each wave owns 2 row-tiles x
// 8 col-tiles of 16x16, processed as two passes of 4 col-tiles (acc[2][4]).
// ---------------------------------------------------------------------------
__global__ __launch_bounds__(256)
void fused_mlp_kernel(const float* __restrict__ obs,       // B x 256 fp32
                      const float* __restrict__ price,     // B x 512 fp32
                      const float* __restrict__ wtab,      // 4 x 512 fp32 (EWMA)
                      const unsigned short* __restrict__ W1,   // 1024 x 288 bf16
                      const unsigned short* __restrict__ W2,   // 1024 x 1024 bf16
                      const float* __restrict__ fc1_b,
                      const float* __restrict__ fc2_b,
                      const float* __restrict__ g1, const float* __restrict__ b1,
                      const float* __restrict__ g2, const float* __restrict__ b2,
                      const float* __restrict__ mean_w,
                      const float* __restrict__ mean_b,
                      const float* __restrict__ log_std,
                      float* __restrict__ out) {
    __shared__ unsigned short Xs[ROWS * XS_STRIDE];   // 18,944 B
    __shared__ unsigned short Hs[ROWS * HS];          // 66,048 B  (h1)
    __shared__ unsigned short Ho[ROWS * HS];          // 66,048 B  (h2)
    __shared__ float ps[256], pq[256];                //  2,048 B
    __shared__ float row_mu[ROWS], row_rs[ROWS];      //    256 B

    const int tid  = threadIdx.x;
    const int lane = tid & 31;
    const int wv   = tid >> 5;          // wave id 0..7
    const int hsel = lane >> 4;         // half-wave select
    const int ln   = lane & 15;
    const int r0   = blockIdx.x * ROWS; // global row base

    // ---------------- prologue: obs -> bf16 Xs, EWMA feats -----------------
    // EWMA weight table aliases Ho (Ho is not written until GEMM2 output).
    float* wls = (float*)Ho;            // 4*512 fp32 = 8 KB
    for (int i = tid; i < NEWMA * Tlen; i += 256) wls[i] = wtab[i];

    for (int i = tid; i < ROWS * OBS; i += 256) {     // coalesced obs pack
        int r = i >> 8, c = i & (OBS - 1);
        Xs[r * XS_STRIDE + c] = f2bf(obs[(size_t)(r0 + r) * OBS + c]);
    }
    for (int i = tid; i < ROWS * (K1P - INDIM); i += 256) {  // zero pad
        int r = i / (K1P - INDIM), c = INDIM + i % (K1P - INDIM);
        Xs[r * XS_STRIDE + c] = 0;
    }
    __syncthreads();

    // one wave per row, 4 rows per wave
    #pragma unroll
    for (int rr = 0; rr < ROWS / 8; ++rr) {
        const int r = wv * (ROWS / 8) + rr;
        const float* pr = price + (size_t)(r0 + r) * Tlen;
        float s0 = 0.f, s1 = 0.f, s2 = 0.f, s3 = 0.f;
        #pragma unroll
        for (int i = 0; i < Tlen / 32; ++i) {
            int t = i * 32 + lane;                    // coalesced
            float p = pr[t];
            s0 += p * wls[0 * Tlen + t];
            s1 += p * wls[1 * Tlen + t];
            s2 += p * wls[2 * Tlen + t];
            s3 += p * wls[3 * Tlen + t];
        }
        #pragma unroll
        for (int off = 16; off > 0; off >>= 1) {      // wave32 butterfly
            s0 += __shfl_xor(s0, off, 32);
            s1 += __shfl_xor(s1, off, 32);
            s2 += __shfl_xor(s2, off, 32);
            s3 += __shfl_xor(s3, off, 32);
        }
        if (lane == 0) {
            float last = pr[Tlen - 1];
            float inv  = 1.0f / (fabsf(last) + 1e-8f);
            Xs[r * XS_STRIDE + OBS + 0] = f2bf((s0 - last) * inv);
            Xs[r * XS_STRIDE + OBS + 1] = f2bf((s1 - last) * inv);
            Xs[r * XS_STRIDE + OBS + 2] = f2bf((s2 - last) * inv);
            Xs[r * XS_STRIDE + OBS + 3] = f2bf((s3 - last) * inv);
        }
    }
    __syncthreads();

    const v8f vzero = {0.f, 0.f, 0.f, 0.f, 0.f, 0.f, 0.f, 0.f};

    // =================== GEMM1: (32 x 288) @ (288 x 1024) ===================
    // two column passes of 4 tiles each; every B fragment feeds 2 WMMAs
    for (int cp = 0; cp < 2; ++cp) {
        v8f acc[2][4];
        #pragma unroll
        for (int rt = 0; rt < 2; ++rt)
            #pragma unroll
            for (int t = 0; t < 4; ++t) acc[rt][t] = vzero;

        for (int kk = 0; kk < K1P / 32; ++kk) {
            const int k0 = kk * 32;
            const unsigned short* ap0 = &Xs[ln * XS_STRIDE + k0 + hsel * 8];
            const unsigned short* ap1 = ap0 + 16 * XS_STRIDE;
            FragU a0, a1;
            a0.q[0] = *(const uint4*)(ap0); a0.q[1] = *(const uint4*)(ap0 + 16);
            a1.q[0] = *(const uint4*)(ap1); a1.q[1] = *(const uint4*)(ap1 + 16);
            #pragma unroll
            for (int t = 0; t < 4; ++t) {
                const int n0 = (wv * 8 + cp * 4 + t) * 16;
                const unsigned short* bp =
                    &W1[(size_t)(n0 + ln) * K1P + k0 + hsel * 8];
                __builtin_prefetch(bp + 32, 0, 1);    // global_prefetch_b8
                FragU b;
                b.q[0] = *(const uint4*)(bp);
                b.q[1] = *(const uint4*)(bp + 16);
                acc[0][t] = __builtin_amdgcn_wmma_f32_16x16x32_bf16(
                    false, a0.v, false, b.v, (short)0, acc[0][t], false, false);
                acc[1][t] = __builtin_amdgcn_wmma_f32_16x16x32_bf16(
                    false, a1.v, false, b.v, (short)0, acc[1][t], false, false);
            }
        }
        // bias + store pre-LN bf16 into Hs (each wave owns disjoint columns)
        #pragma unroll
        for (int t = 0; t < 4; ++t) {
            const int n = (wv * 8 + cp * 4 + t) * 16 + ln;
            const float bias = fc1_b[n];
            #pragma unroll
            for (int rt = 0; rt < 2; ++rt)
                #pragma unroll
                for (int j = 0; j < 8; ++j) {
                    const int m = rt * 16 + j + hsel * 8;
                    Hs[m * HS + n] = f2bf(acc[rt][t][j] + bias);
                }
        }
    }
    __syncthreads();

    // ---- LayerNorm1 + ReLU (8 threads per row, 128 elems each) ----
    {
        const int row = tid >> 3, ch = tid & 7;
        float s = 0.f, q = 0.f;
        #pragma unroll 8
        for (int i = 0; i < 128; ++i) {
            float v = bf2f(Hs[row * HS + ch * 128 + i]);
            s += v; q += v * v;
        }
        ps[tid] = s; pq[tid] = q;
        __syncthreads();
        if (ch == 0) {
            float S = 0.f, Q = 0.f;
            #pragma unroll
            for (int i = 0; i < 8; ++i) { S += ps[row * 8 + i]; Q += pq[row * 8 + i]; }
            float mu  = S * (1.0f / HID);
            float var = Q * (1.0f / HID) - mu * mu;
            row_mu[row] = mu;
            row_rs[row] = rsqrtf(var + LN_EPS);
        }
        __syncthreads();
        const float mu = row_mu[row], rs = row_rs[row];
        #pragma unroll 8
        for (int i = 0; i < 128; ++i) {
            const int n = ch * 128 + i;
            float v = bf2f(Hs[row * HS + n]);
            v = (v - mu) * rs * g1[n] + b1[n];
            v = fmaxf(v, 0.0f);
            Hs[row * HS + n] = f2bf(v);
        }
    }
    __syncthreads();     // wls (aliasing Ho) dead from here; Ho writable

    // =================== GEMM2: (32 x 1024) @ (1024 x 1024) =================
    for (int cp = 0; cp < 2; ++cp) {
        v8f acc[2][4];
        #pragma unroll
        for (int rt = 0; rt < 2; ++rt)
            #pragma unroll
            for (int t = 0; t < 4; ++t) acc[rt][t] = vzero;

        for (int kk = 0; kk < HID / 32; ++kk) {
            const int k0 = kk * 32;
            const unsigned short* ap0 = &Hs[ln * HS + k0 + hsel * 8];
            const unsigned short* ap1 = ap0 + 16 * HS;
            FragU a0, a1;
            a0.q[0] = *(const uint4*)(ap0); a0.q[1] = *(const uint4*)(ap0 + 16);
            a1.q[0] = *(const uint4*)(ap1); a1.q[1] = *(const uint4*)(ap1 + 16);
            #pragma unroll
            for (int t = 0; t < 4; ++t) {
                const int n0 = (wv * 8 + cp * 4 + t) * 16;
                const unsigned short* bp =
                    &W2[(size_t)(n0 + ln) * HID + k0 + hsel * 8];
                __builtin_prefetch(bp + 32, 0, 1);
                FragU b;
                b.q[0] = *(const uint4*)(bp);
                b.q[1] = *(const uint4*)(bp + 16);
                acc[0][t] = __builtin_amdgcn_wmma_f32_16x16x32_bf16(
                    false, a0.v, false, b.v, (short)0, acc[0][t], false, false);
                acc[1][t] = __builtin_amdgcn_wmma_f32_16x16x32_bf16(
                    false, a1.v, false, b.v, (short)0, acc[1][t], false, false);
            }
        }
        // write h2 into Ho (separate buffer: Hs still being read by others)
        #pragma unroll
        for (int t = 0; t < 4; ++t) {
            const int n = (wv * 8 + cp * 4 + t) * 16 + ln;
            const float bias = fc2_b[n];
            #pragma unroll
            for (int rt = 0; rt < 2; ++rt)
                #pragma unroll
                for (int j = 0; j < 8; ++j) {
                    const int m = rt * 16 + j + hsel * 8;
                    Ho[m * HS + n] = f2bf(acc[rt][t][j] + bias);
                }
        }
    }
    __syncthreads();

    // ---- LayerNorm2 + ReLU + head dot ----
    {
        const int row = tid >> 3, ch = tid & 7;
        float s = 0.f, q = 0.f;
        #pragma unroll 8
        for (int i = 0; i < 128; ++i) {
            float v = bf2f(Ho[row * HS + ch * 128 + i]);
            s += v; q += v * v;
        }
        ps[tid] = s; pq[tid] = q;
        __syncthreads();
        if (ch == 0) {
            float S = 0.f, Q = 0.f;
            #pragma unroll
            for (int i = 0; i < 8; ++i) { S += ps[row * 8 + i]; Q += pq[row * 8 + i]; }
            float mu  = S * (1.0f / HID);
            float var = Q * (1.0f / HID) - mu * mu;
            row_mu[row] = mu;
            row_rs[row] = rsqrtf(var + LN_EPS);
        }
        __syncthreads();
        const float mu = row_mu[row], rs = row_rs[row];
        float d = 0.f;
        #pragma unroll 8
        for (int i = 0; i < 128; ++i) {
            const int n = ch * 128 + i;
            float v = bf2f(Ho[row * HS + n]);
            v = (v - mu) * rs * g2[n] + b2[n];
            v = fmaxf(v, 0.0f);
            d += v * mean_w[n];               // fold head dot-product in
        }
        ps[tid] = d;
        __syncthreads();
        if (ch == 0) {
            float D = 0.f;
            #pragma unroll
            for (int i = 0; i < 8; ++i) D += ps[row * 8 + i];
            const int b = r0 + row;
            out[b] = tanhf(D + mean_b[0]);
            float sd = expf(log_std[0]);
            sd = fminf(fmaxf(sd, 0.01f), 0.5f);
            out[Btot + b] = sd;
        }
    }
}

// ---------------------------------------------------------------------------
// Launch
// ---------------------------------------------------------------------------
extern "C" void kernel_launch(void* const* d_in, const int* in_sizes, int n_in,
                              void* d_out, int out_size, void* d_ws, size_t ws_size,
                              hipStream_t stream) {
    const float* obs      = (const float*)d_in[0];
    const float* price    = (const float*)d_in[1];
    const float* log_hl   = (const float*)d_in[2];
    const float* fc1_w    = (const float*)d_in[3];
    const float* fc1_b    = (const float*)d_in[4];
    const float* fc2_w    = (const float*)d_in[5];
    const float* fc2_b    = (const float*)d_in[6];
    const float* ln1_g    = (const float*)d_in[7];
    const float* ln1_b    = (const float*)d_in[8];
    const float* ln2_g    = (const float*)d_in[9];
    const float* ln2_b    = (const float*)d_in[10];
    const float* mean_w   = (const float*)d_in[11];
    const float* mean_b   = (const float*)d_in[12];
    const float* log_std  = (const float*)d_in[13];
    float* out = (float*)d_out;

    // workspace carve-up (256B aligned)
    char* ws = (char*)d_ws;
    float*          w_tab = (float*)(ws);                            //   8 KB
    unsigned short* W1b   = (unsigned short*)(ws + 8192);            // 576 KB
    unsigned short* W2b   = (unsigned short*)(ws + 8192 + 589824);   //   2 MB

    ewma_weights_kernel<<<1, 512, 0, stream>>>(log_hl, w_tab);
    convert_w1_kernel<<<(HID * K1P) / 256, 256, 0, stream>>>(fc1_w, W1b);
    convert_w2_kernel<<<(HID * HID) / 256, 256, 0, stream>>>(fc2_w, W2b);
    fused_mlp_kernel<<<Btot / ROWS, 256, 0, stream>>>(
        obs, price, w_tab, W1b, W2b, fc1_b, fc2_b, ln1_g, ln1_b, ln2_g, ln2_b,
        mean_w, mean_b, log_std, out);
}
